// HSA_14774687498387
// MI455X (gfx1250) — matrix operand order repaired and checked by reading, not compiled, not run on hardware
//
#include <hip/hip_runtime.h>
#include <hip/hip_bf16.h>
#include <math.h>

// ---------------------------------------------------------------------------
// MI455X / gfx1250 implementation. wave32, WMMA f16 (16x16x32) everywhere.
// Inputs flattened from setup_inputs() dict order (83 arrays) -- see index
// map in kernel_launch.
// ---------------------------------------------------------------------------

#define B_   64
#define F_   256
#define D_   256
#define K_   8
#define H_   4
#define HD_  64
#define PF_  512
#define NROW (B_ * F_)   // 16384
#define SROW (B_ * K_)   // 512

typedef _Float16 v16h __attribute__((ext_vector_type(16)));
typedef _Float16 v8h  __attribute__((ext_vector_type(8)));
typedef float    v8f  __attribute__((ext_vector_type(8)));

// ---------------------------------------------------------------------------
// WMMA helpers (contiguous k = 16*(lane/16) + s in both A and B, wave32)
// ---------------------------------------------------------------------------
__device__ __forceinline__ v8f wmma16(v16h a, v16h b, v8f c) {
  // emits v_wmma_f32_16x16x32_f16
  return __builtin_amdgcn_wmma_f32_16x16x32_f16(
      /*neg_a=*/false, a, /*neg_b=*/false, b,
      /*c_mod=*/(short)0, c, /*reuse_a=*/false, /*reuse_b=*/false);
}

// A matrix 16x32 f16: lane l (0..15): M=l, K = s..s+15, s=0;
// lane 16..31: M=l-16, s=16.  Row-major source [rows x ld].
__device__ __forceinline__ v16h load_fragA(const _Float16* p, int ld, int row0, int k0) {
  int lane = threadIdx.x & 31;
  const _Float16* base = p + (size_t)(row0 + (lane & 15)) * ld + k0 + ((lane >> 4) << 4);
  v8h lo = *(const v8h*)(base);
  v8h hi = *(const v8h*)(base + 8);
  v16h r;
#pragma unroll
  for (int i = 0; i < 8; ++i) { r[i] = lo[i]; r[i + 8] = hi[i]; }
  return r;
}

// B matrix 32x16 f16: lane l: N = l&15, K = (l>=16 ? 16 : 0) + i (16 contiguous).
// Source rows are the N dimension (weights W[N,K] row-major, K contiguous).
__device__ __forceinline__ v16h load_fragB(const _Float16* p, int ld, int n0, int k0) {
  int lane = threadIdx.x & 31;
  const _Float16* base = p + (size_t)(n0 + (lane & 15)) * ld + k0 + ((lane >> 4) << 4);
  v8h lo = *(const v8h*)(base);
  v8h hi = *(const v8h*)(base + 8);
  v16h r;
#pragma unroll
  for (int i = 0; i < 8; ++i) { r[i] = lo[i]; r[i + 8] = hi[i]; }
  return r;
}

// ---------------------------------------------------------------------------
// Generic GEMM: C[M,N] = alpha*(A[M,K] @ W[N,K]^T) + bias + resid ; ReLU opt.
// Writes f32 (Cf) and/or f16 (Ch). M%64==0, N%64==0, K%32==0.
// Block 256 thr = 8 waves; block tile 64x64; wave: rows (w&3)*16, cols (w>>2)*32.
// ---------------------------------------------------------------------------
__global__ __launch_bounds__(256)
void k_gemm(const _Float16* __restrict__ A, const _Float16* __restrict__ W,
            const float* __restrict__ bias, const float* __restrict__ resid,
            float* __restrict__ Cf, _Float16* __restrict__ Ch,
            int M, int N, int K, float alpha, int relu) {
  int w    = threadIdx.x >> 5;
  int lane = threadIdx.x & 31;
  int m0 = blockIdx.x * 64 + (w & 3) * 16;
  int n0 = blockIdx.y * 64 + (w >> 2) * 32;
  if (m0 >= M || n0 >= N) return;

  v8f acc0 = {};
  v8f acc1 = {};
  for (int kk = 0; kk < K; kk += 32) {
    v16h a  = load_fragA(A, K, m0, kk);
    v16h b0 = load_fragB(W, K, n0, kk);
    acc0 = wmma16(a, b0, acc0);
    v16h b1 = load_fragB(W, K, n0 + 16, kk);
    acc1 = wmma16(a, b1, acc1);
  }

  int colc = lane & 15;
  int ro   = (lane >> 4) << 3;
#pragma unroll
  for (int t = 0; t < 2; ++t) {
    v8f acc = t ? acc1 : acc0;
    int nc  = n0 + t * 16 + colc;
#pragma unroll
    for (int r = 0; r < 8; ++r) {
      int row = m0 + ro + r;
      float v = acc[r] * alpha;
      if (bias)  v += bias[nc];
      if (resid) v += resid[(size_t)row * N + nc];
      if (relu)  v = fmaxf(v, 0.0f);
      size_t idx = (size_t)row * N + nc;
      if (Cf) Cf[idx] = v;
      if (Ch) Ch[idx] = (_Float16)v;
    }
  }
}

// ---------------------------------------------------------------------------
// Fused MHA: one wave per (b, h, 16-query tile). q,k,v,o are f16 [B,F,D]
// with head slice cols [h*64, h*64+64). scale 1/sqrt(64) = 0.125.
// ---------------------------------------------------------------------------
__global__ __launch_bounds__(32)
void k_attn(const _Float16* __restrict__ q, const _Float16* __restrict__ k,
            const _Float16* __restrict__ v, _Float16* __restrict__ o) {
  __shared__ float sc[16][260];
  __shared__ float red[32], red2[32];

  int b = blockIdx.z, h = blockIdx.y;
  int q0 = blockIdx.x * 16;
  int hoff = h * HD_;
  int lane = threadIdx.x & 31;
  int colc = lane & 15;
  int ro   = (lane >> 4) << 3;
  const size_t bo = (size_t)b * F_ * D_;

  // ---- S = (Q K^T) * 0.125 ----
  v16h qa0 = load_fragA(q + bo + hoff, D_, q0, 0);
  v16h qa1 = load_fragA(q + bo + hoff, D_, q0, 32);
  for (int jt = 0; jt < F_ / 16; ++jt) {
    int n0 = jt * 16;
    v8f acc = {};
    v16h kb0 = load_fragB(k + bo + hoff, D_, n0, 0);
    acc = wmma16(qa0, kb0, acc);
    v16h kb1 = load_fragB(k + bo + hoff, D_, n0, 32);
    acc = wmma16(qa1, kb1, acc);
#pragma unroll
    for (int r = 0; r < 8; ++r) sc[ro + r][n0 + colc] = acc[r] * 0.125f;
  }
  __syncthreads();

  // ---- softmax over 256 keys per row; 2 lanes per row ----
  int part = lane >> 4, rw = lane & 15, c0 = part * 128;
  float mx = -1e30f;
  for (int c = 0; c < 128; ++c) mx = fmaxf(mx, sc[rw][c0 + c]);
  red[lane] = mx;
  __syncthreads();
  float m = fmaxf(red[rw], red[rw + 16]);
  float ssum = 0.f;
  for (int c = 0; c < 128; ++c) {
    float e = __expf(sc[rw][c0 + c] - m);
    sc[rw][c0 + c] = e;
    ssum += e;
  }
  red2[lane] = ssum;
  __syncthreads();
  float inv = 1.f / (red2[rw] + red2[rw + 16]);
  for (int c = 0; c < 128; ++c) sc[rw][c0 + c] *= inv;
  __syncthreads();

  // ---- O = P V ----
  v8f accO[4] = {{}, {}, {}, {}};
  for (int kc = 0; kc < F_ / 32; ++kc) {
    int kk = kc * 32;
    v16h pf;
    int rp = lane & 15, s = (lane >> 4) << 4;
#pragma unroll
    for (int i = 0; i < 8; ++i) {
      pf[i]     = (_Float16)sc[rp][kk + s + i];
      pf[i + 8] = (_Float16)sc[rp][kk + 8 + s + i];
    }
#pragma unroll
    for (int ct = 0; ct < 4; ++ct) {
      v16h vf;
      int coln  = hoff + ct * 16 + (lane & 15);
      int kbase = kk + ((lane >> 4) << 4);
      const _Float16* vp = v + bo + (size_t)kbase * D_ + coln;
#pragma unroll
      for (int i = 0; i < 16; ++i) vf[i] = vp[(size_t)i * D_];
      accO[ct] = wmma16(pf, vf, accO[ct]);
    }
  }
#pragma unroll
  for (int ct = 0; ct < 4; ++ct) {
#pragma unroll
    for (int r = 0; r < 8; ++r) {
      int row = q0 + ro + r;
      int col = hoff + ct * 16 + colc;
      o[bo + (size_t)row * D_ + col] = (_Float16)accO[ct][r];
    }
  }
}

// ---------------------------------------------------------------------------
// Row LayerNorm over D=256: y = (x-mu)*rsqrt(var+1e-5)*g + b ; f32/f16 outs.
// ---------------------------------------------------------------------------
__global__ __launch_bounds__(256)
void k_ln(const float* __restrict__ x, const float* __restrict__ g,
          const float* __restrict__ b, float* __restrict__ yf,
          _Float16* __restrict__ yh) {
  __shared__ float s1[256], s2[256];
  int row = blockIdx.x, t = threadIdx.x;
  size_t idx = (size_t)row * 256 + t;
  float v = x[idx];
  s1[t] = v; s2[t] = v * v;
  __syncthreads();
  for (int st = 128; st > 0; st >>= 1) {
    if (t < st) { s1[t] += s1[t + st]; s2[t] += s2[t + st]; }
    __syncthreads();
  }
  float mean = s1[0] * (1.f / 256.f);
  float var  = s2[0] * (1.f / 256.f) - mean * mean;
  float rs   = rsqrtf(var + 1e-5f);
  float o = (v - mean) * rs * g[t] + b[t];
  if (yf) yf[idx] = o;
  if (yh) yh[idx] = (_Float16)o;
}

// ---------------------------------------------------------------------------
// x*sqrt(D) + pos ; f32 + f16 outs.  n = B*F*D, pos broadcast over batch.
// ---------------------------------------------------------------------------
__global__ void k_scale_pos(const float* __restrict__ x, const float* __restrict__ pos,
                            float* __restrict__ yf, _Float16* __restrict__ yh, size_t n) {
  size_t i = (size_t)blockIdx.x * 256 + threadIdx.x;
  if (i >= n) return;
  float v = x[i] * 16.0f + pos[i & (size_t)(F_ * D_ - 1)];
  yf[i] = v; yh[i] = (_Float16)v;
}

__global__ void k_cvt(const float* __restrict__ in, _Float16* __restrict__ out, size_t n) {
  size_t i = (size_t)blockIdx.x * 256 + threadIdx.x;
  if (i < n) out[i] = (_Float16)in[i];
}

__global__ void k_slot_init(const float* __restrict__ noise, const float* __restrict__ mu,
                            const float* __restrict__ ls, float* __restrict__ slots, size_t n) {
  size_t i = (size_t)blockIdx.x * 256 + threadIdx.x;
  if (i >= n) return;
  int d = (int)(i & (D_ - 1));
  slots[i] = mu[d] + __expf(ls[d]) * noise[i];
}

// ---------------------------------------------------------------------------
// Slot-attention update: per (b,h): logits = k·q, softmax over 8 slots,
// +EPS, normalize over 256 inputs, upd = a^T v. k,v f16 [B,F,D]; q f32 [B,K,D].
// ---------------------------------------------------------------------------
__global__ __launch_bounds__(256)
void k_slot_update(const _Float16* __restrict__ kbuf, const _Float16* __restrict__ vbuf,
                   const float* __restrict__ qbuf, float* __restrict__ upd) {
  __shared__ _Float16 vsh[256][HD_];   // 32 KB
  __shared__ float    qsh[K_][HD_];    // 2 KB
  __shared__ float    ash[256][K_];    // 8 KB
  __shared__ float    csum[K_];
  int h = blockIdx.x, b = blockIdx.y, t = threadIdx.x;
  int hoff = h * HD_;

  for (int i = t; i < 256 * HD_; i += 256) {
    int n = i >> 6, d = i & 63;
    vsh[n][d] = vbuf[((size_t)(b * F_ + n)) * D_ + hoff + d];
  }
  for (int i = t; i < K_ * HD_; i += 256) {
    int s = i >> 6, d = i & 63;
    qsh[s][d] = qbuf[((size_t)(b * K_ + s)) * D_ + hoff + d];
  }
  __syncthreads();

  const _Float16* krow = kbuf + ((size_t)(b * F_ + t)) * D_ + hoff;
  float lg[K_];
#pragma unroll
  for (int s = 0; s < K_; ++s) lg[s] = 0.f;
  for (int d = 0; d < HD_; ++d) {
    float kv = (float)krow[d];
#pragma unroll
    for (int s = 0; s < K_; ++s) lg[s] += kv * qsh[s][d];
  }
  float m = lg[0];
#pragma unroll
  for (int s = 1; s < K_; ++s) m = fmaxf(m, lg[s]);
  float sum = 0.f;
#pragma unroll
  for (int s = 0; s < K_; ++s) { lg[s] = __expf(lg[s] - m); sum += lg[s]; }
  float inv = 1.f / sum;
#pragma unroll
  for (int s = 0; s < K_; ++s) ash[t][s] = lg[s] * inv + 1e-8f;
  __syncthreads();

  if (t < K_) {
    float s0 = 0.f;
    for (int n = 0; n < 256; ++n) s0 += ash[n][t];
    csum[t] = s0;
  }
  __syncthreads();

  for (int o = t; o < K_ * HD_; o += 256) {
    int s = o >> 6, d = o & 63;
    float acc = 0.f;
    for (int n = 0; n < 256; ++n) acc += ash[n][s] * (float)vsh[n][d];
    upd[((size_t)(b * K_ + s)) * D_ + hoff + d] = acc / csum[s];
  }
}

// ---------------------------------------------------------------------------
// GRUCell (torch gate order r,z,n): gi/gh [SROW,768]; hio [SROW,256] in-place.
// ---------------------------------------------------------------------------
__global__ __launch_bounds__(256)
void k_gru(const float* __restrict__ gi, const float* __restrict__ gh,
           float* __restrict__ hio) {
  int row = blockIdx.x, d = threadIdx.x;
  size_t gb = (size_t)row * 768;
  float ir = gi[gb + d],       hr = gh[gb + d];
  float iz = gi[gb + 256 + d], hz = gh[gb + 256 + d];
  float in_ = gi[gb + 512 + d], hn = gh[gb + 512 + d];
  float r = 1.f / (1.f + __expf(-(ir + hr)));
  float z = 1.f / (1.f + __expf(-(iz + hz)));
  float n = tanhf(in_ + r * hn);
  size_t idx = (size_t)row * 256 + d;
  float hp = hio[idx];
  hio[idx] = (1.f - z) * n + z * hp;
}

// ---------------------------------------------------------------------------
// Fused decoder: per block = one (b,k) slot, 32 broadcast positions.
// x = slot + pos[f]; LN; h = relu(x @ W1^T + b1) via WMMA; out = h @ W2^T + b2.
// out [B,K,F,2] f32. W1 f16 [256,256]; W2 f32 [2,256].
// ---------------------------------------------------------------------------
__global__ __launch_bounds__(256)
void k_dec(const float* __restrict__ slots, const float* __restrict__ pos,
           const float* __restrict__ g, const float* __restrict__ bb,
           const _Float16* __restrict__ W1, const float* __restrict__ b1,
           const float* __restrict__ W2, const float* __restrict__ b2,
           float* __restrict__ out) {
  __shared__ float    sl[256];
  __shared__ _Float16 Ash[32][256];    // 16 KB
  __shared__ _Float16 Hsh[32][256];    // 16 KB
  __shared__ float    redm[32][8], redv[32][8];
  __shared__ float    mrow[32], vrow[32];

  int bk = blockIdx.y;
  int f0 = blockIdx.x * 32;
  int t  = threadIdx.x;

  sl[t] = slots[(size_t)bk * 256 + t];
  __syncthreads();

  // ---- broadcast + LN (8 threads per row, 32 elems each) ----
  int r = t >> 3, gs = t & 7;
  const float* prow = pos + (size_t)(f0 + r) * 256;
  float ps = 0.f, pss = 0.f;
  for (int j = 0; j < 32; ++j) {
    int d = gs * 32 + j;
    float v = sl[d] + prow[d];
    ps += v; pss += v * v;
  }
  redm[r][gs] = ps; redv[r][gs] = pss;
  __syncthreads();
  if (gs == 0) {
    float a = 0.f, c = 0.f;
    for (int j = 0; j < 8; ++j) { a += redm[r][j]; c += redv[r][j]; }
    float mean = a * (1.f / 256.f);
    mrow[r] = mean;
    vrow[r] = c * (1.f / 256.f) - mean * mean;
  }
  __syncthreads();
  float mean = mrow[r];
  float rs   = rsqrtf(vrow[r] + 1e-5f);
  for (int j = 0; j < 32; ++j) {
    int d = gs * 32 + j;
    float v = sl[d] + prow[d];
    Ash[r][d] = (_Float16)((v - mean) * rs * g[d] + bb[d]);
  }
  __syncthreads();

  // ---- hidden = relu(A @ W1^T + b1) via WMMA ----
  int w = t >> 5, lane = t & 31;
  int rt = (w & 1) * 16;
  int cq = w >> 1;  // 0..3, each owns 4 col tiles
  int colc = lane & 15, ro = (lane >> 4) << 3;
  for (int ci = 0; ci < 4; ++ci) {
    int n0 = (cq * 4 + ci) * 16;
    v8f acc = {};
    for (int kk = 0; kk < 256; kk += 32) {
      v16h a  = load_fragA(&Ash[0][0], 256, rt, kk);
      v16h bf = load_fragB(W1, 256, n0, kk);
      acc = wmma16(a, bf, acc);
    }
#pragma unroll
    for (int rr = 0; rr < 8; ++rr) {
      int row = rt + ro + rr;
      int col = n0 + colc;
      float v = fmaxf(acc[rr] + b1[col], 0.0f);
      Hsh[row][col] = (_Float16)v;
    }
  }
  __syncthreads();

  // ---- out[row, c] = Hsh[row] . W2[c] + b2[c] ----
  if (t < 64) {
    int row = t >> 1, c = t & 1;
    float acc = b2[c];
    const float* w2 = W2 + c * 256;
    for (int d = 0; d < 256; ++d) acc += (float)Hsh[row][d] * w2[d];
    out[((size_t)bk * F_ + f0 + row) * 2 + c] = acc;
  }
}

// ---------------------------------------------------------------------------
// Host orchestration
// ---------------------------------------------------------------------------
extern "C" void kernel_launch(void* const* d_in, const int* in_sizes, int n_in,
                              void* d_out, int out_size, void* d_ws, size_t ws_size,
                              hipStream_t stream) {
  (void)in_sizes; (void)n_in; (void)out_size; (void)ws_size;
  // ---- input index map (setup_inputs dict order, recursively flattened) ----
  // 0 x[B,F,D]  1 slots_noise[B,K,D]  2 enc.pos[F,D]
  // 3+16*blk : Wq bq Wk bk Wv bv Wo bo g1 b1 g2 b2 W1 bf1 W2 bf2   (blk=0..2)
  // 51 sae.g 52 sae.b 53 sae.W1 54 sae.b1 55 sae.W2 56 sae.b2 57 mu 58 log_sigma
  // 59 msa.Wq 60 msa.Wk 61 msa.Wv 62 gin 63 bin 64 gs 65 bs 66 gm 67 bm
  // 68 Wih 69 Whh 70 bih 71 bhh 72 msa.W1 73 msa.b1 74 msa.W2 75 msa.b2
  // 76 dec.pos 77 dec.g 78 dec.b 79 dec.W1 80 dec.b1 81 dec.W2 82 dec.b2
  auto F32 = [&](int i) { return (const float*)d_in[i]; };

  // ---- workspace layout ----
  char*  wsb = (char*)d_ws;
  size_t off = 0;
  auto alloc = [&](size_t bytes) -> char* {
    char* p = wsb + off;
    off = (off + bytes + 255) & ~(size_t)255;
    return p;
  };
  float*     Xf  = (float*)alloc((size_t)NROW * D_ * 4);
  _Float16*  Xb  = (_Float16*)alloc((size_t)NROW * D_ * 2);
  float*     Tf  = (float*)alloc((size_t)NROW * D_ * 4);
  _Float16*  Qb  = (_Float16*)alloc((size_t)NROW * D_ * 2);
  _Float16*  Kb  = (_Float16*)alloc((size_t)NROW * D_ * 2);
  _Float16*  Vb  = (_Float16*)alloc((size_t)NROW * D_ * 2);
  _Float16*  Ob  = (_Float16*)alloc((size_t)NROW * D_ * 2);
  _Float16*  Hb  = (_Float16*)alloc((size_t)NROW * PF_ * 2);
  _Float16*  Wh  = (_Float16*)alloc((size_t)2752512 * 2);   // f16 weight pool
  float*     slotsF = (float*)alloc((size_t)SROW * D_ * 4);
  _Float16*  sBh    = (_Float16*)alloc((size_t)SROW * D_ * 2);
  _Float16*  uBh    = (_Float16*)alloc((size_t)SROW * D_ * 2);
  float*     updF   = (float*)alloc((size_t)SROW * D_ * 4);
  float*     qsF    = (float*)alloc((size_t)SROW * D_ * 4);
  float*     giF    = (float*)alloc((size_t)SROW * 768 * 4);
  float*     ghF    = (float*)alloc((size_t)SROW * 768 * 4);
  _Float16*  mlpHh  = (_Float16*)alloc((size_t)SROW * PF_ * 2);

  // ---- convert all GEMM weights to f16 once ----
  size_t woff = 0;
  auto wcvt = [&](int idx, size_t n) -> _Float16* {
    _Float16* dst = Wh + woff;
    woff += n;
    k_cvt<<<dim3((unsigned)((n + 255) / 256)), 256, 0, stream>>>(F32(idx), dst, n);
    return dst;
  };
  _Float16 *eWq[3], *eWk[3], *eWv[3], *eWo[3], *eW1[3], *eW2[3];
  for (int blk = 0; blk < 3; ++blk) {
    int base = 3 + blk * 16;
    eWq[blk] = wcvt(base + 0,  65536);
    eWk[blk] = wcvt(base + 2,  65536);
    eWv[blk] = wcvt(base + 4,  65536);
    eWo[blk] = wcvt(base + 6,  65536);
    eW1[blk] = wcvt(base + 12, 131072);
    eW2[blk] = wcvt(base + 14, 131072);
  }
  _Float16* saeW1 = wcvt(53, 131072);
  _Float16* saeW2 = wcvt(55, 131072);
  _Float16* msaWq = wcvt(59, 65536);
  _Float16* msaWk = wcvt(60, 65536);
  _Float16* msaWv = wcvt(61, 65536);
  _Float16* gWih  = wcvt(68, 196608);
  _Float16* gWhh  = wcvt(69, 196608);
  _Float16* msaW1 = wcvt(72, 131072);
  _Float16* msaW2 = wcvt(74, 131072);
  _Float16* decW1 = wcvt(79, 65536);

  auto gemm = [&](const _Float16* A, const _Float16* W, const float* bias,
                  const float* resid, float* Cf, _Float16* Ch,
                  int M, int N, int Kd, float alpha, int relu) {
    k_gemm<<<dim3(M / 64, N / 64), 256, 0, stream>>>(A, W, bias, resid, Cf, Ch,
                                                     M, N, Kd, alpha, relu);
  };

  // ================= encoder =================
  {
    size_t n = (size_t)NROW * D_;
    k_scale_pos<<<dim3((unsigned)((n + 255) / 256)), 256, 0, stream>>>(
        F32(0), F32(2), Xf, Xb, n);
  }
  for (int blk = 0; blk < 3; ++blk) {
    int base = 3 + blk * 16;
    gemm(Xb, eWq[blk], F32(base + 1), nullptr, nullptr, Qb, NROW, D_, D_, 1.f, 0);
    gemm(Xb, eWk[blk], F32(base + 3), nullptr, nullptr, Kb, NROW, D_, D_, 1.f, 0);
    gemm(Xb, eWv[blk], F32(base + 5), nullptr, nullptr, Vb, NROW, D_, D_, 1.f, 0);
    k_attn<<<dim3(F_ / 16, H_, B_), 32, 0, stream>>>(Qb, Kb, Vb, Ob);
    gemm(Ob, eWo[blk], F32(base + 7), Xf, Tf, nullptr, NROW, D_, D_, 1.f, 0);
    k_ln<<<NROW, 256, 0, stream>>>(Tf, F32(base + 8), F32(base + 9), Xf, Xb);
    gemm(Xb, eW1[blk], F32(base + 13), nullptr, nullptr, Hb, NROW, PF_, D_, 1.f, 1);
    gemm(Hb, eW2[blk], F32(base + 15), Xf, Tf, nullptr, NROW, D_, PF_, 1.f, 0);
    k_ln<<<NROW, 256, 0, stream>>>(Tf, F32(base + 10), F32(base + 11), Xf, Xb);
  }

  // ================= SAE front MLP =================
  k_ln<<<NROW, 256, 0, stream>>>(Xf, F32(51), F32(52), nullptr, Xb);
  gemm(Xb, saeW1, F32(54), nullptr, nullptr, Hb, NROW, PF_, D_, 1.f, 1);
  gemm(Hb, saeW2, F32(56), nullptr, Xf, nullptr, NROW, D_, PF_, 1.f, 0);

  // slots = mu + exp(log_sigma) * noise
  {
    size_t n = (size_t)SROW * D_;
    k_slot_init<<<dim3((unsigned)((n + 255) / 256)), 256, 0, stream>>>(
        F32(1), F32(57), F32(58), slotsF, n);
  }

  // ================= slot attention =================
  k_ln<<<NROW, 256, 0, stream>>>(Xf, F32(62), F32(63), nullptr, Xb);
  gemm(Xb, msaWk, nullptr, nullptr, nullptr, Kb, NROW, D_, D_, 0.125f, 0);  // k/sqrt(HD)
  gemm(Xb, msaWv, nullptr, nullptr, nullptr, Vb, NROW, D_, D_, 1.f, 0);

  for (int it = 0; it < 3; ++it) {
    k_ln<<<SROW, 256, 0, stream>>>(slotsF, F32(64), F32(65), nullptr, sBh);
    gemm(sBh, msaWq, nullptr, nullptr, qsF, nullptr, SROW, D_, D_, 1.f, 0);
    k_slot_update<<<dim3(H_, B_), 256, 0, stream>>>(Kb, Vb, qsF, updF);
    {
      size_t n = (size_t)SROW * D_;
      k_cvt<<<dim3((unsigned)((n + 255) / 256)), 256, 0, stream>>>(updF, uBh, n);
      k_cvt<<<dim3((unsigned)((n + 255) / 256)), 256, 0, stream>>>(slotsF, sBh, n);
    }
    gemm(uBh, gWih, F32(70), nullptr, giF, nullptr, SROW, 768, D_, 1.f, 0);
    gemm(sBh, gWhh, F32(71), nullptr, ghF, nullptr, SROW, 768, D_, 1.f, 0);
    k_gru<<<SROW, 256, 0, stream>>>(giF, ghF, slotsF);
    k_ln<<<SROW, 256, 0, stream>>>(slotsF, F32(66), F32(67), nullptr, sBh);
    gemm(sBh, msaW1, F32(73), nullptr, nullptr, mlpHh, SROW, PF_, D_, 1.f, 1);
    gemm(mlpHh, msaW2, F32(75), slotsF, slotsF, nullptr, SROW, D_, PF_, 1.f, 0);
  }

  // ================= decoder (fused) =================
  k_dec<<<dim3(F_ / 32, B_ * K_), 256, 0, stream>>>(
      slotsF, F32(76), F32(77), F32(78), decW1, F32(80), F32(81), F32(82),
      (float*)d_out);
}
